// PrincipleNet_89352499626642
// MI455X (gfx1250) — compile-verified, matrix operand order checked by reading, not run on hardware
//
#include <hip/hip_runtime.h>
#include <hip/hip_bf16.h>
#include <stdint.h>

// ---------------------------------------------------------------------------
// PrincipleNet forward for MI455X (gfx1250, wave32, WMMA).
// All convs run as implicit GEMM on v_wmma_f32_16x16x32_bf16 (f32 accumulate).
// Activations live in NHWC bf16 with a 1-pixel zero halo (130x130xC).
// ---------------------------------------------------------------------------

typedef __attribute__((ext_vector_type(16))) __bf16 bf16x16;
typedef __attribute__((ext_vector_type(8)))  float  f32x8;

#define B_    4
#define H_    128
#define W_    128
#define P_    (H_ * W_)          // 16384 pixels per image
#define PH_   130                // padded height
#define PW_   130                // padded width
#define OPS_  6
#define KPN_  5

// Pair->K mapping for 16-bit A/B fragments (ISA 7.12.2, 16-bit A 16x32):
// lanes 0-15 hold K pairs {0,1},{2,3},{4,5},{6,7},{16,17},... lanes 16-31 are +8.
__device__ __forceinline__ int kmap(int p, int half) {
  return 2 * p + ((p >= 4) ? 8 : 0) + 8 * half;
}

__device__ __forceinline__ uint32_t ld32(const __bf16* p) {
  return *reinterpret_cast<const uint32_t*>(p);
}

__device__ __forceinline__ void put2(bf16x16& f, int p, uint32_t u) {
  union { uint32_t u; __bf16 h[2]; } cv;
  cv.u = u;
  f[2 * p]     = cv.h[0];
  f[2 * p + 1] = cv.h[1];
}

// ---------------------------------------------------------------------------
// Utility kernels
// ---------------------------------------------------------------------------

__global__ void fill_zero_u32(uint32_t* __restrict__ p, size_t n) {
  size_t i = (size_t)blockIdx.x * blockDim.x + threadIdx.x;
  if (i < n) p[i] = 0u;
}

// OIHW f32 -> [CO][ks2*CI] bf16 with K ordered (spatial, ci) so K-pairs are
// channel pairs (contiguous in NHWC activations too).
__global__ void repack_w(const float* __restrict__ w, __bf16* __restrict__ out,
                         int CO, int CI, int KS2) {
  int i = blockIdx.x * blockDim.x + threadIdx.x;
  int total = CO * CI * KS2;
  if (i >= total) return;
  int m = i / (CI * KS2);
  int r = i - m * (CI * KS2);
  int s = r / CI;
  int ci = r - s * CI;
  out[i] = (__bf16)w[(size_t)((m * CI + ci) * KS2) + s];
}

// NCHW f32 -> NHWC bf16 (unpadded)
__global__ void cvt_nchw_to_nhwc_bf16(const float* __restrict__ in,
                                      __bf16* __restrict__ out, int C) {
  size_t i = (size_t)blockIdx.x * blockDim.x + threadIdx.x;
  size_t total = (size_t)B_ * C * P_;
  if (i >= total) return;
  int p = (int)(i % P_);
  int c = (int)((i / P_) % C);
  int b = (int)(i / ((size_t)C * P_));
  out[((size_t)b * P_ + p) * C + c] = (__bf16)in[i];
}

// ---------------------------------------------------------------------------
// Routing head: global-avg-pool -> 1x1 -> 1x1 -> softmax  (tiny, one block/b)
// ---------------------------------------------------------------------------
__global__ void __launch_bounds__(64)
pred_kernel(const float* __restrict__ img,
            const float* __restrict__ w1, const float* __restrict__ b1,
            const float* __restrict__ w2, const float* __restrict__ b2,
            float* __restrict__ pred_out, float* __restrict__ mixw) {
  __shared__ float g[64], p1s[64], p2s[OPS_];
  const int b = blockIdx.x, c = threadIdx.x;
  const float* src = img + ((size_t)b * 64 + c) * P_;
  float s = 0.f;
  for (int i = 0; i < P_; ++i) s += src[i];
  g[c] = s * (1.0f / P_);
  __syncthreads();
  float a = b1[c];
  for (int k = 0; k < 64; ++k) a += w1[c * 64 + k] * g[k];
  p1s[c] = a;
  __syncthreads();
  if (c < OPS_) {
    float v = b2[c];
    for (int k = 0; k < 64; ++k) v += w2[c * 64 + k] * p1s[k];
    p2s[c] = v;
  }
  __syncthreads();
  if (c == 0) {
    float mx = p2s[0];
    for (int o = 1; o < OPS_; ++o) mx = fmaxf(mx, p2s[o]);
    float e[OPS_], den = 0.f;
    for (int o = 0; o < OPS_; ++o) { e[o] = __expf(p2s[o] - mx); den += e[o]; }
    for (int o = 0; o < OPS_; ++o) {
      float pr = e[o] / den;
      pred_out[b * OPS_ + o] = pr;
      p2s[o] = pr;
    }
    mixw[b * 3 + 0] = p2s[2] + p2s[4];  // wp (point)
    mixw[b * 3 + 1] = p2s[0] + p2s[1];  // wa (add)
    mixw[b * 3 + 2] = p2s[3];           // wc (convolve)
  }
}

// ---------------------------------------------------------------------------
// 1x1 conv 64->64 via WMMA, writing interleaved into padded dmap (ch 2m+off),
// optionally also a dense f32 NHWC copy. One wave: 32 Cout x 64 pixels.
// ---------------------------------------------------------------------------
__global__ void __launch_bounds__(32)
mlp1x1_wmma(const __bf16* __restrict__ in_bf,   // [B][P][64]
            const __bf16* __restrict__ wpack,   // [64][64]
            const float*  __restrict__ bias,    // [64]
            __bf16* __restrict__ dmap_pad,      // [B][PH][PW][128]
            float*  __restrict__ out_f32,       // optional [B][P][64]
            int chan_off) {
  const int lane = threadIdx.x & 31, half = lane >> 4, ln = lane & 15;
  const int b = blockIdx.z, mb = blockIdx.y * 32, pxb = blockIdx.x * 64;

  int py[4], pxx[4];
  const __bf16* pbase[4];
  #pragma unroll
  for (int j = 0; j < 4; ++j) {
    int px = pxb + j * 16 + ln;
    py[j] = px >> 7; pxx[j] = px & (W_ - 1);
    pbase[j] = in_bf + ((size_t)b * P_ + px) * 64;
  }

  f32x8 acc[2][4];
  #pragma unroll
  for (int i = 0; i < 2; ++i)
    #pragma unroll
    for (int j = 0; j < 4; ++j)
      #pragma unroll
      for (int v = 0; v < 8; ++v) acc[i][j][v] = 0.f;

  #pragma unroll
  for (int kb = 0; kb < 64; kb += 32) {
    bf16x16 af[2];
    #pragma unroll
    for (int mi = 0; mi < 2; ++mi) {
      const __bf16* wr = wpack + (size_t)(mb + mi * 16 + ln) * 64 + kb;
      #pragma unroll
      for (int p = 0; p < 8; ++p) put2(af[mi], p, ld32(wr + kmap(p, half)));
    }
    bf16x16 bfg[4];
    #pragma unroll
    for (int j = 0; j < 4; ++j)
      #pragma unroll
      for (int p = 0; p < 8; ++p)
        put2(bfg[j], p, ld32(pbase[j] + kb + kmap(p, half)));
    #pragma unroll
    for (int mi = 0; mi < 2; ++mi)
      #pragma unroll
      for (int j = 0; j < 4; ++j)
        acc[mi][j] = __builtin_amdgcn_wmma_f32_16x16x32_bf16(
            false, af[mi], false, bfg[j], (short)0, acc[mi][j], false, false);
  }

  #pragma unroll
  for (int mi = 0; mi < 2; ++mi)
    #pragma unroll
    for (int v = 0; v < 8; ++v) {
      const int m = mb + mi * 16 + v + 8 * half;
      const float bv = bias[m];
      #pragma unroll
      for (int j = 0; j < 4; ++j) {
        float val = acc[mi][j][v] + bv;
        const size_t pint = ((size_t)b * PH_ + (py[j] + 1)) * PW_ + (pxx[j] + 1);
        dmap_pad[pint * 128 + 2 * m + chan_off] = (__bf16)val;
        if (out_f32)
          out_f32[((size_t)b * P_ + (py[j] << 7) + pxx[j]) * 64 + m] = val;
      }
    }
}

// ---------------------------------------------------------------------------
// Generic SAME 3x3 conv via implicit-GEMM WMMA.
// One wave computes a 32(Cout) x 64(pixel) tile; K = 9*CIN in steps of 32.
// Fused bias + optional ReLU + optional residual (bf16 padded) +
// optional extra f32 add; writes bf16 padded and/or f32 NHWC.
// ---------------------------------------------------------------------------
template <int CIN, int COUT, bool RELU>
__global__ void __launch_bounds__(32)
conv3x3_wmma(const __bf16* __restrict__ in_pad,    // [B][PH][PW][CIN]
             const __bf16* __restrict__ wpack,     // [COUT][9*CIN]
             const float*  __restrict__ bias,      // [COUT]
             const __bf16* __restrict__ res_pad,   // opt [B][PH][PW][COUT]
             const float*  __restrict__ extra_f32, // opt [B][P][COUT]
             __bf16* __restrict__ out_pad,         // opt [B][PH][PW][COUT]
             float*  __restrict__ out_f32) {       // opt [B][P][COUT]
  constexpr int K = 9 * CIN;
  constexpr int MASK = CIN - 1;
  const int lane = threadIdx.x & 31, half = lane >> 4, ln = lane & 15;
  const int b = blockIdx.z, mb = blockIdx.y * 32, pxb = blockIdx.x * 64;

  int py[4], pxx[4];
  #pragma unroll
  for (int j = 0; j < 4; ++j) {
    int px = pxb + j * 16 + ln;
    py[j] = px >> 7; pxx[j] = px & (W_ - 1);
  }

  f32x8 acc[2][4];
  #pragma unroll
  for (int i = 0; i < 2; ++i)
    #pragma unroll
    for (int j = 0; j < 4; ++j)
      #pragma unroll
      for (int v = 0; v < 8; ++v) acc[i][j][v] = 0.f;

  const __bf16* inb = in_pad + (size_t)b * PH_ * PW_ * CIN;

  for (int kb = 0; kb < K; kb += 32) {
    // A fragments: weight rows
    bf16x16 af[2];
    #pragma unroll
    for (int mi = 0; mi < 2; ++mi) {
      const __bf16* wr = wpack + (size_t)(mb + mi * 16 + ln) * K + kb;
      #pragma unroll
      for (int p = 0; p < 8; ++p) put2(af[mi], p, ld32(wr + kmap(p, half)));
      if (kb + 32 < K) __builtin_prefetch(wr + 32, 0, 0);  // global_prefetch_b8
    }
    // decode K -> (tap dy,dx, ci) once per pair slot
    int di8[8], dj8[8], ci8[8];
    #pragma unroll
    for (int p = 0; p < 8; ++p) {
      int kk = kb + kmap(p, half);
      int s  = kk / CIN;          // CIN pow2 -> shift
      ci8[p] = kk & MASK;
      di8[p] = s / 3;
      dj8[p] = s - 3 * di8[p];
    }
    // B fragments: input pixels (halo makes every tap in-bounds)
    bf16x16 bfg[4];
    #pragma unroll
    for (int j = 0; j < 4; ++j)
      #pragma unroll
      for (int p = 0; p < 8; ++p) {
        const __bf16* src =
            inb + ((size_t)(py[j] + di8[p]) * PW_ + (pxx[j] + dj8[p])) * CIN + ci8[p];
        put2(bfg[j], p, ld32(src));
      }
    #pragma unroll
    for (int mi = 0; mi < 2; ++mi)
      #pragma unroll
      for (int j = 0; j < 4; ++j)
        acc[mi][j] = __builtin_amdgcn_wmma_f32_16x16x32_bf16(
            false, af[mi], false, bfg[j], (short)0, acc[mi][j], false, false);
  }

  #pragma unroll
  for (int mi = 0; mi < 2; ++mi)
    #pragma unroll
    for (int v = 0; v < 8; ++v) {
      const int m = mb + mi * 16 + v + 8 * half;
      const float bv = bias[m];
      #pragma unroll
      for (int j = 0; j < 4; ++j) {
        float val = acc[mi][j][v] + bv;
        if (RELU) val = val > 0.f ? val : 0.f;
        const int y = py[j], x = pxx[j];
        const size_t pint = ((size_t)b * PH_ + (y + 1)) * PW_ + (x + 1);
        const size_t pf = (size_t)b * P_ + (y << 7) + x;
        if (res_pad)   val += (float)res_pad[pint * COUT + m];
        if (extra_f32) val += extra_f32[pf * COUT + m];
        if (out_pad)   out_pad[pint * COUT + m] = (__bf16)val;
        if (out_f32)   out_f32[pf * COUT + m] = val;
      }
    }
}

// ---------------------------------------------------------------------------
// Fused kernel-prediction (1x1 conv 64->1600, WMMA into LDS) + per-pixel 5x5
// depthwise combine with edge clamping. 4 waves / 16 pixels per block.
// ---------------------------------------------------------------------------
__global__ void __launch_bounds__(128)
kpn_fused(const __bf16* __restrict__ h_pad,   // [B][PH][PW][64]
          const __bf16* __restrict__ kwp,     // [1600][64]
          const float*  __restrict__ kbias,   // [1600]
          const float*  __restrict__ imgm,    // [B][P][64]
          float* __restrict__ out3) {         // [B][P][64]
  __shared__ __bf16 bk[1600 * 16];            // 51200 B of LDS
  const int tid = threadIdx.x;
  const int wave = tid >> 5, lane = tid & 31;
  const int half = lane >> 4, ln = lane & 15;
  const int b = blockIdx.y;
  const int pxb = blockIdx.x * 16;
  const int px = pxb + ln;
  const int y = px >> 7, x = px & (W_ - 1);

  // B fragments (K=64) for our 16 pixels (same for every M tile)
  bf16x16 bfg[2];
  const __bf16* hp = h_pad + (((size_t)b * PH_ + (y + 1)) * PW_ + (x + 1)) * 64;
  #pragma unroll
  for (int kb = 0; kb < 2; ++kb)
    #pragma unroll
    for (int p = 0; p < 8; ++p)
      put2(bfg[kb], p, ld32(hp + kb * 32 + kmap(p, half)));

  for (int t = 0; t < 25; ++t) {              // each wave: 25 M-tiles (400 rows)
    const int mbase = (wave * 25 + t) * 16;
    f32x8 acc;
    #pragma unroll
    for (int v = 0; v < 8; ++v) acc[v] = 0.f;
    #pragma unroll
    for (int kb = 0; kb < 2; ++kb) {
      bf16x16 af;
      const __bf16* wr = kwp + (size_t)(mbase + ln) * 64 + kb * 32;
      #pragma unroll
      for (int p = 0; p < 8; ++p) put2(af, p, ld32(wr + kmap(p, half)));
      acc = __builtin_amdgcn_wmma_f32_16x16x32_bf16(
          false, af, false, bfg[kb], (short)0, acc, false, false);
    }
    #pragma unroll
    for (int v = 0; v < 8; ++v) {
      const int row = mbase + v + 8 * half;
      bk[row * 16 + ln] = (__bf16)(acc[v] + kbias[row]);
    }
  }
  __syncthreads();

  // depthwise combine: 64 channels x 16 pixels, 8 tasks per thread
  for (int task = tid; task < 64 * 16; task += 128) {
    const int c = task >> 4, pl = task & 15;
    const int ppx = pxb + pl;
    const int y0 = ppx >> 7, x0 = ppx & (W_ - 1);
    float sum = 0.f;
    #pragma unroll
    for (int i = 0; i < KPN_; ++i) {
      int yy = y0 + i - 2; yy = yy < 0 ? 0 : (yy > H_ - 1 ? H_ - 1 : yy);
      #pragma unroll
      for (int j = 0; j < KPN_; ++j) {
        int xx = x0 + j - 2; xx = xx < 0 ? 0 : (xx > W_ - 1 ? W_ - 1 : xx);
        sum += (float)bk[(c * 25 + i * 5 + j) * 16 + pl] *
               imgm[((size_t)b * P_ + (yy << 7) + xx) * 64 + c];
      }
    }
    out3[((size_t)b * P_ + ppx) * 64 + c] = sum;
  }
}

// ---------------------------------------------------------------------------
// Final soft mixture -> NCHW f32 output
// ---------------------------------------------------------------------------
__global__ void final_mix(const float* __restrict__ imgm,
                          const float* __restrict__ out3,
                          const float* __restrict__ wgt,   // [B][P][128]
                          const float* __restrict__ mixw,  // [B][3]
                          float* __restrict__ out) {
  size_t i = (size_t)blockIdx.x * blockDim.x + threadIdx.x;
  if (i >= (size_t)B_ * 64 * P_) return;
  int p = (int)(i % P_);
  int c = (int)((i / P_) % 64);
  int b = (int)(i / ((size_t)64 * P_));
  size_t pf = (size_t)b * P_ + p;
  float imm   = imgm[pf * 64 + c];
  float gamma = wgt[pf * 128 + c];
  float beta  = wgt[pf * 128 + 64 + c];
  float o3    = out3[pf * 64 + c];
  float wp = mixw[b * 3 + 0], wa = mixw[b * 3 + 1], wc = mixw[b * 3 + 2];
  out[i] = wp * (gamma * imm + beta) + wa * imm + wc * o3;
}

// ---------------------------------------------------------------------------
// Host orchestration
// ---------------------------------------------------------------------------
extern "C" void kernel_launch(void* const* d_in, const int* in_sizes, int n_in,
                              void* d_out, int out_size, void* d_ws, size_t ws_size,
                              hipStream_t stream) {
  (void)in_sizes; (void)n_in; (void)out_size; (void)ws_size;
  const float* img   = (const float*)d_in[0];
  const float* degra = (const float*)d_in[1];
  // param leaves, jax pytree order (sorted dict keys, 'b' before 'w'):
  const float* add_c0_b   = (const float*)d_in[2];
  const float* add_c0_w   = (const float*)d_in[3];
  const float* add_r1c1_b = (const float*)d_in[4];
  const float* add_r1c1_w = (const float*)d_in[5];
  const float* add_r1c2_b = (const float*)d_in[6];
  const float* add_r1c2_w = (const float*)d_in[7];
  const float* add_r2c1_b = (const float*)d_in[8];
  const float* add_r2c1_w = (const float*)d_in[9];
  const float* add_r2c2_b = (const float*)d_in[10];
  const float* add_r2c2_w = (const float*)d_in[11];
  const float* cv_c0_b    = (const float*)d_in[12];
  const float* cv_c0_w    = (const float*)d_in[13];
  const float* cv_k_b     = (const float*)d_in[14];
  const float* cv_k_w     = (const float*)d_in[15];
  const float* cv_r1c1_b  = (const float*)d_in[16];
  const float* cv_r1c1_w  = (const float*)d_in[17];
  const float* cv_r1c2_b  = (const float*)d_in[18];
  const float* cv_r1c2_w  = (const float*)d_in[19];
  const float* cv_r2c1_b  = (const float*)d_in[20];
  const float* cv_r2c1_w  = (const float*)d_in[21];
  const float* cv_r2c2_b  = (const float*)d_in[22];
  const float* cv_r2c2_w  = (const float*)d_in[23];
  const float* mlpd_b     = (const float*)d_in[24];
  const float* mlpd_w     = (const float*)d_in[25];
  const float* mlpi_b     = (const float*)d_in[26];
  const float* mlpi_w     = (const float*)d_in[27];
  const float* pt_c0_b    = (const float*)d_in[28];
  const float* pt_c0_w    = (const float*)d_in[29];
  const float* pt_r1c1_b  = (const float*)d_in[30];
  const float* pt_r1c1_w  = (const float*)d_in[31];
  const float* pt_r1c2_b  = (const float*)d_in[32];
  const float* pt_r1c2_w  = (const float*)d_in[33];
  const float* pt_r2c1_b  = (const float*)d_in[34];
  const float* pt_r2c1_w  = (const float*)d_in[35];
  const float* pt_r2c2_b  = (const float*)d_in[36];
  const float* pt_r2c2_w  = (const float*)d_in[37];
  const float* pred1_b    = (const float*)d_in[38];
  const float* pred1_w    = (const float*)d_in[39];
  const float* pred2_b    = (const float*)d_in[40];
  const float* pred2_w    = (const float*)d_in[41];

  // workspace carving
  char* wsb = (char*)d_ws;
  size_t off = 0;
  auto alloc = [&](size_t bytes) -> void* {
    void* p = wsb + off;
    off = (off + bytes + 255) & ~(size_t)255;
    return p;
  };
  const size_t NHWC64  = (size_t)B_ * P_ * 64;
  const size_t NHWC128 = (size_t)B_ * P_ * 128;
  const size_t PAD64   = (size_t)B_ * PH_ * PW_ * 64;
  const size_t PAD128  = (size_t)B_ * PH_ * PW_ * 128;

  __bf16* imgbf   = (__bf16*)alloc(NHWC64 * 2);
  __bf16* degbf   = (__bf16*)alloc(NHWC64 * 2);
  __bf16* dmap    = (__bf16*)alloc(PAD128 * 2);
  __bf16* p64a    = (__bf16*)alloc(PAD64 * 2);
  __bf16* p64b    = (__bf16*)alloc(PAD64 * 2);
  __bf16* p128a   = (__bf16*)alloc(PAD128 * 2);
  __bf16* p128b   = (__bf16*)alloc(PAD128 * 2);
  float*  img2f   = (float*)alloc(NHWC64 * 4);
  float*  imgmf   = (float*)alloc(NHWC64 * 4);
  float*  out3f   = (float*)alloc(NHWC64 * 4);
  float*  wgtf    = (float*)alloc(NHWC128 * 4);
  float*  mixw    = (float*)alloc(B_ * 3 * 4);
  __bf16* pwMlpI  = (__bf16*)alloc(64 * 64 * 2);
  __bf16* pwMlpD  = (__bf16*)alloc(64 * 64 * 2);
  __bf16* pwAddC0 = (__bf16*)alloc(64 * 1152 * 2);
  __bf16* pwAddR[4]; for (int i = 0; i < 4; ++i) pwAddR[i] = (__bf16*)alloc(64 * 576 * 2);
  __bf16* pwCvC0  = (__bf16*)alloc(64 * 1152 * 2);
  __bf16* pwCvR[4]; for (int i = 0; i < 4; ++i) pwCvR[i] = (__bf16*)alloc(64 * 576 * 2);
  __bf16* pwK     = (__bf16*)alloc(1600 * 64 * 2);
  __bf16* pwPtC0  = (__bf16*)alloc(128 * 1152 * 2);
  __bf16* pwPtR[4]; for (int i = 0; i < 4; ++i) pwPtR[i] = (__bf16*)alloc(128 * 1152 * 2);

  // zero halos (whole padded buffers; interiors are rewritten every launch)
  auto zfill = [&](void* p, size_t bytes) {
    size_t n = bytes / 4;
    fill_zero_u32<<<(unsigned)((n + 255) / 256), 256, 0, stream>>>((uint32_t*)p, n);
  };
  zfill(dmap,  PAD128 * 2);
  zfill(p64a,  PAD64 * 2);
  zfill(p64b,  PAD64 * 2);
  zfill(p128a, PAD128 * 2);
  zfill(p128b, PAD128 * 2);

  // weight repacks
  auto rep = [&](const float* w, __bf16* o, int CO, int CI, int KS2) {
    int tot = CO * CI * KS2;
    repack_w<<<(tot + 255) / 256, 256, 0, stream>>>(w, o, CO, CI, KS2);
  };
  rep(mlpi_w, pwMlpI, 64, 64, 1);
  rep(mlpd_w, pwMlpD, 64, 64, 1);
  rep(add_c0_w, pwAddC0, 64, 128, 9);
  rep(add_r1c1_w, pwAddR[0], 64, 64, 9);
  rep(add_r1c2_w, pwAddR[1], 64, 64, 9);
  rep(add_r2c1_w, pwAddR[2], 64, 64, 9);
  rep(add_r2c2_w, pwAddR[3], 64, 64, 9);
  rep(cv_c0_w, pwCvC0, 64, 128, 9);
  rep(cv_r1c1_w, pwCvR[0], 64, 64, 9);
  rep(cv_r1c2_w, pwCvR[1], 64, 64, 9);
  rep(cv_r2c1_w, pwCvR[2], 64, 64, 9);
  rep(cv_r2c2_w, pwCvR[3], 64, 64, 9);
  rep(cv_k_w, pwK, 1600, 64, 1);
  rep(pt_c0_w, pwPtC0, 128, 128, 9);
  rep(pt_r1c1_w, pwPtR[0], 128, 128, 9);
  rep(pt_r1c2_w, pwPtR[1], 128, 128, 9);
  rep(pt_r2c1_w, pwPtR[2], 128, 128, 9);
  rep(pt_r2c2_w, pwPtR[3], 128, 128, 9);

  // NCHW f32 -> NHWC bf16
  {
    size_t tot = (size_t)B_ * 64 * P_;
    cvt_nchw_to_nhwc_bf16<<<(unsigned)((tot + 255) / 256), 256, 0, stream>>>(img, imgbf, 64);
    cvt_nchw_to_nhwc_bf16<<<(unsigned)((tot + 255) / 256), 256, 0, stream>>>(degra, degbf, 64);
  }

  // routing head (on raw img, before mlp)
  float* pred_out = (float*)d_out + (size_t)B_ * 64 * P_;
  pred_kernel<<<B_, 64, 0, stream>>>(img, pred1_w, pred1_b, pred2_w, pred2_b,
                                     pred_out, mixw);

  // mlp 1x1 convs -> interleaved dmap (img2 also kept in f32 for imgm)
  {
    dim3 g(P_ / 64, 2, B_);
    mlp1x1_wmma<<<g, 32, 0, stream>>>(imgbf, pwMlpI, mlpi_b, dmap, img2f, 0);
    mlp1x1_wmma<<<g, 32, 0, stream>>>(degbf, pwMlpD, mlpd_b, dmap, nullptr, 1);
  }

  dim3 g64(P_ / 64, 2, B_);    // Cout=64 convs
  dim3 g128(P_ / 64, 4, B_);   // Cout=128 convs

  // add route: deg = res(res(conv(dmap))) ; imgm = img2 + deg (fused)
  conv3x3_wmma<128, 64, false><<<g64, 32, 0, stream>>>(dmap, pwAddC0, add_c0_b,
      nullptr, nullptr, p64a, nullptr);
  conv3x3_wmma<64, 64, true><<<g64, 32, 0, stream>>>(p64a, pwAddR[0], add_r1c1_b,
      nullptr, nullptr, p64b, nullptr);
  conv3x3_wmma<64, 64, false><<<g64, 32, 0, stream>>>(p64b, pwAddR[1], add_r1c2_b,
      p64a, nullptr, p64a, nullptr);
  conv3x3_wmma<64, 64, true><<<g64, 32, 0, stream>>>(p64a, pwAddR[2], add_r2c1_b,
      nullptr, nullptr, p64b, nullptr);
  conv3x3_wmma<64, 64, false><<<g64, 32, 0, stream>>>(p64b, pwAddR[3], add_r2c2_b,
      p64a, img2f, nullptr, imgmf);

  // convolve route: h = res(res(conv(dmap))) -> fused kpn on imgm
  conv3x3_wmma<128, 64, false><<<g64, 32, 0, stream>>>(dmap, pwCvC0, cv_c0_b,
      nullptr, nullptr, p64a, nullptr);
  conv3x3_wmma<64, 64, true><<<g64, 32, 0, stream>>>(p64a, pwCvR[0], cv_r1c1_b,
      nullptr, nullptr, p64b, nullptr);
  conv3x3_wmma<64, 64, false><<<g64, 32, 0, stream>>>(p64b, pwCvR[1], cv_r1c2_b,
      p64a, nullptr, p64a, nullptr);
  conv3x3_wmma<64, 64, true><<<g64, 32, 0, stream>>>(p64a, pwCvR[2], cv_r2c1_b,
      nullptr, nullptr, p64b, nullptr);
  conv3x3_wmma<64, 64, false><<<g64, 32, 0, stream>>>(p64b, pwCvR[3], cv_r2c2_b,
      p64a, nullptr, p64a, nullptr);
  {
    dim3 gk(P_ / 16, B_, 1);
    kpn_fused<<<gk, 128, 0, stream>>>(p64a, pwK, cv_k_b, imgmf, out3f);
  }

  // point route: wgt = res(res(conv(dmap))) at 128 ch -> f32
  conv3x3_wmma<128, 128, false><<<g128, 32, 0, stream>>>(dmap, pwPtC0, pt_c0_b,
      nullptr, nullptr, p128a, nullptr);
  conv3x3_wmma<128, 128, true><<<g128, 32, 0, stream>>>(p128a, pwPtR[0], pt_r1c1_b,
      nullptr, nullptr, p128b, nullptr);
  conv3x3_wmma<128, 128, false><<<g128, 32, 0, stream>>>(p128b, pwPtR[1], pt_r1c2_b,
      p128a, nullptr, p128a, nullptr);
  conv3x3_wmma<128, 128, true><<<g128, 32, 0, stream>>>(p128a, pwPtR[2], pt_r2c1_b,
      nullptr, nullptr, p128b, nullptr);
  conv3x3_wmma<128, 128, false><<<g128, 32, 0, stream>>>(p128b, pwPtR[3], pt_r2c2_b,
      p128a, nullptr, nullptr, wgtf);

  // final soft mixture -> NCHW out
  {
    size_t tot = (size_t)B_ * 64 * P_;
    final_mix<<<(unsigned)((tot + 255) / 256), 256, 0, stream>>>(
        imgmf, out3f, wgtf, mixw, (float*)d_out);
  }
}
